// TTOA_25022479466799
// MI455X (gfx1250) — matrix operand order repaired and checked by reading, not compiled, not run on hardware
//
#include <hip/hip_runtime.h>
#include <math.h>
#include <stdint.h>

typedef __attribute__((ext_vector_type(16))) _Float16 v16h;
typedef __attribute__((ext_vector_type(8)))  _Float16 v8h;
typedef __attribute__((ext_vector_type(8)))  float    v8f;

#define B_    4
#define C_    64
#define H_    256
#define W_    256
#define HW_   (H_ * W_)
#define NPIX  (B_ * HW_)          // 262144 pixels
#define NT    (NPIX / 16)         // 16384 16-pixel tiles

__device__ __forceinline__ float sigm(float x) { return 1.0f / (1.0f + __expf(-x)); }

__device__ __forceinline__ v8f wmma16(v16h a, v16h b, v8f c) {
  // D = A(16x32 f16) * B(32x16 f16) + C(16x16 f32)
  return __builtin_amdgcn_wmma_f32_16x16x32_f16(false, a, false, b, (short)0, c, false, false);
}

// raw LDS byte offset of a __shared__ pointer (generic addr low 32 bits == LDS offset)
__device__ __forceinline__ unsigned lds_off(const void* p) {
  return (unsigned)(uintptr_t)p;
}

// ---------------------------------------------------------------------------
// Kernel 1: fused preprocess  (1x1 conv 64->32 -> 1x1 conv 32->64)
// in : NCHW f32, out: NHWC f32 (workspace; L2-resident for later gathers)
// One wave per 16-pixel tile. blockIdx.y selects low/high tensor.
// Input tile is staged to LDS with async copies (ASYNCcnt path), then the
// WMMA A fragments are assembled from LDS.
// ---------------------------------------------------------------------------
__global__ __launch_bounds__(128) void k_pre(
    const float* __restrict__ xlo, const float* __restrict__ xhi,
    const float* __restrict__ w1, const float* __restrict__ b1,   // (32,64)
    const float* __restrict__ w2, const float* __restrict__ b2,   // (64,32)
    float* __restrict__ plo, float* __restrict__ phi)
{
  const float* x = blockIdx.y ? xhi : xlo;
  float*       p = blockIdx.y ? phi : plo;

  const int wave = threadIdx.x >> 5;
  const int lane = threadIdx.x & 31;
  const int colx = lane & 15;       // A row / B column within 16
  const int hi   = lane >> 4;       // lane half selects K sub-block

  const long P0 = ((long)blockIdx.x * 4 + wave) * 16;   // tile base pixel
  const int  b  = (int)(P0 / HW_);
  const int  s0 = (int)(P0 % HW_);
  const float* xb = x + (size_t)b * C_ * HW_;

  __shared__ __align__(16) float    xs[4][64 * 16];     // [wave][ch][px] staged f32 tile
  __shared__ __align__(16) _Float16 hbuf[4][16 * 32];   // [wave][px][hidden ch]
  float*    xt = xs[wave];
  _Float16* hl = hbuf[wave];

  // ---- async-stage the 16px x 64ch f32 input tile into LDS (4KB/wave) ----
  {
    const unsigned lbase = lds_off(xt);
    #pragma unroll
    for (int it = 0; it < 8; ++it) {
      int q  = it * 32 + lane;          // 16B chunk id, 256 per tile
      int K  = q >> 2;                  // input channel (64B row per channel)
      int bo = (q & 3) * 16;            // byte offset within the 64B row
      const char* ga = (const char*)(xb + (size_t)K * HW_ + s0) + bo;
      unsigned lo = lbase + (unsigned)q * 16;
      asm volatile("global_load_async_to_lds_b128 %0, %1, off"
                   :: "v"(lo), "v"(ga) : "memory");
    }
    asm volatile("s_wait_asynccnt 0" ::: "memory");
  }

  // ---- GEMM1: [16 pix x 64 in] * [64 x 32] -> hidden (f32 acc) ----
  v8f h0 = {}, h1 = {};
  #pragma unroll
  for (int ks = 0; ks < 2; ++ks) {
    v16h a;
    #pragma unroll
    for (int e = 0; e < 16; ++e) {
      int K = ks * 32 + (e >> 3) * 16 + hi * 8 + (e & 7);   // ISA A f16 16x32 layout
      a[e] = (_Float16)xt[K * 16 + colx];
    }
    v16h bm0, bm1;
    #pragma unroll
    for (int j = 0; j < 16; ++j) {
      int K = ks * 32 + (j >> 3) * 16 + hi * 8 + (j & 7);   // ISA B layout: same K order as A
      bm0[j] = (_Float16)w1[(size_t)colx * 64 + K];
      bm1[j] = (_Float16)w1[(size_t)(colx + 16) * 64 + K];
    }
    h0 = wmma16(a, bm0, h0);
    h1 = wmma16(a, bm1, h1);
  }

  // bias + stash hidden as f16 [pixel][32] in LDS (wave-private region)
  {
    float bA = b1[colx], bB = b1[colx + 16];
    #pragma unroll
    for (int r = 0; r < 8; ++r) {
      int m = r + hi * 8;                                   // D: VGPR r -> pixel r / r+8
      hl[m * 32 + colx]      = (_Float16)(h0[r] + bA);
      hl[m * 32 + colx + 16] = (_Float16)(h1[r] + bB);
    }
  }
  asm volatile("s_wait_dscnt 0" ::: "memory");              // LDS RAW fence (split counters)

  // ---- GEMM2: [16 x 32 hidden] * [32 x 64] -> out ----
  v16h a2;
  {
    v8h lo = *(const v8h*)&hl[colx * 32 + hi * 8];
    v8h up = *(const v8h*)&hl[colx * 32 + 16 + hi * 8];
    #pragma unroll
    for (int e = 0; e < 8; ++e) { a2[e] = lo[e]; a2[8 + e] = up[e]; }
  }
  #pragma unroll
  for (int nt = 0; nt < 4; ++nt) {
    int n = nt * 16 + colx;
    v16h bm;
    #pragma unroll
    for (int j = 0; j < 16; ++j) bm[j] = (_Float16)w2[(size_t)n * 32 + (j >> 3) * 16 + hi * 8 + (j & 7)];
    v8f o = {};
    o = wmma16(a2, bm, o);
    float bias = b2[n];
    #pragma unroll
    for (int r = 0; r < 8; ++r) {
      int m = r + hi * 8;
      p[(size_t)(P0 + m) * 64 + n] = o[r] + bias;           // NHWC store, 64B coalesced
    }
  }
}

// ---------------------------------------------------------------------------
// Kernel 2: offset conv (3x1 pad(1,0) or 1x3 pad(0,1)), 64 -> 9 channels
// im2col A tile (16 x 192) built from contiguous NHWC rows, GEMM vs padded
// 192x16 weight (cols 9..15 zero).  blockIdx.y: 0 = column conv, 1 = row conv.
// ---------------------------------------------------------------------------
__global__ __launch_bounds__(128) void k_off(
    const float* __restrict__ plo,
    const float* __restrict__ ow_c, const float* __restrict__ ob_c,
    const float* __restrict__ ow_r, const float* __restrict__ ob_r,
    float* __restrict__ off_c, float* __restrict__ off_r)
{
  const int isCol = (blockIdx.y == 0);
  const float* ow  = isCol ? ow_c : ow_r;
  const float* ob  = isCol ? ob_c : ob_r;
  float*       off = isCol ? off_c : off_r;

  const int wave = threadIdx.x >> 5;
  const int lane = threadIdx.x & 31;
  const int colx = lane & 15;
  const int hi   = lane >> 4;

  const long P0 = ((long)blockIdx.x * 4 + wave) * 16;
  const int  b  = (int)(P0 / HW_);
  const int  s0 = (int)(P0 % HW_);
  const int  y  = s0 / W_;
  const int  x0 = s0 % W_;

  __shared__ __align__(16) _Float16 Ab[4][16 * 192];
  _Float16* A = Ab[wave];

  // build A: row p = pixel, K = t*64 + c (zero-padded taps), fully coalesced loads
  for (int p = 0; p < 16; ++p) {
    #pragma unroll
    for (int t = 0; t < 3; ++t) {
      int yy = isCol ? (y + t - 1) : y;
      int xx = isCol ? (x0 + p)    : (x0 + p + t - 1);
      float v0 = 0.f, v1 = 0.f;
      if (yy >= 0 && yy < H_ && xx >= 0 && xx < W_) {
        const float* src = plo + ((size_t)(b * H_ + yy) * W_ + xx) * 64;
        v0 = src[lane];
        v1 = src[lane + 32];
      }
      A[p * 192 + t * 64 + lane]      = (_Float16)v0;
      A[p * 192 + t * 64 + lane + 32] = (_Float16)v1;
    }
  }
  asm volatile("s_wait_dscnt 0" ::: "memory");

  // GEMM: [16 x 192] * [192 x 16(9 live)]
  v8f acc = {};
  #pragma unroll
  for (int ks = 0; ks < 6; ++ks) {
    v16h a;
    {
      v8h lo = *(const v8h*)&A[colx * 192 + ks * 32 + hi * 8];
      v8h up = *(const v8h*)&A[colx * 192 + ks * 32 + 16 + hi * 8];
      #pragma unroll
      for (int e = 0; e < 8; ++e) { a[e] = lo[e]; a[8 + e] = up[e]; }
    }
    v16h bm;
    #pragma unroll
    for (int j = 0; j < 16; ++j) {
      int K = ks * 32 + (j >> 3) * 16 + hi * 8 + (j & 7);
      int t = K / 64, c = K % 64;
      bm[j] = (colx < 9) ? (_Float16)ow[(size_t)(colx * 64 + c) * 3 + t] : (_Float16)0.0f;
    }
    acc = wmma16(a, bm, acc);
  }
  float bias = (colx < 9) ? ob[colx] : 0.f;
  #pragma unroll
  for (int r = 0; r < 8; ++r) {
    int m = r + hi * 8;
    off[(size_t)(P0 + m) * 16 + colx] = acc[r] + bias;      // n: 0-2=dy 3-5=dx 6-8=mask
  }
}

// ---------------------------------------------------------------------------
// Kernel 3: both deformable convs + gated combine.
// One block = one 16-pixel tile; 4 waves each own a 16-out-channel N-tile.
// A tile (16 x 192) = mask * bilinear gathers from NHWC p_low (L2-resident).
// ---------------------------------------------------------------------------
__global__ __launch_bounds__(128) void k_deform(
    const float* __restrict__ plo, const float* __restrict__ phi,
    const float* __restrict__ off_c, const float* __restrict__ off_r,
    const float* __restrict__ wc, const float* __restrict__ bc,
    const float* __restrict__ wr, const float* __restrict__ br,
    float* __restrict__ out)
{
  const int wave = threadIdx.x >> 5;
  const int lane = threadIdx.x & 31;
  const int colx = lane & 15;
  const int hi   = lane >> 4;

  const long P0 = (long)blockIdx.x * 16;
  const int  b  = (int)(P0 / HW_);
  const int  s0 = (int)(P0 % HW_);
  const int  y  = s0 / W_;
  const int  x0 = s0 % W_;
  const int  n  = wave * 16 + colx;          // this lane's output channel

  __shared__ __align__(16) _Float16 A[16 * 192];

  __builtin_prefetch(off_c + P0 * 16, 0, 1);
  __builtin_prefetch(off_r + P0 * 16, 0, 1);

  auto fetch = [&](int yy, int xx, int c) -> float {
    return (yy >= 0 && yy < H_ && xx >= 0 && xx < W_)
           ? plo[((size_t)(b * H_ + yy) * W_ + xx) * 64 + c] : 0.0f;
  };

  v8f dacc[2];

  #pragma unroll
  for (int conv = 0; conv < 2; ++conv) {
    const int isCol = (conv == 0);
    const float* off = isCol ? off_c : off_r;
    const float* w   = isCol ? wc : wr;

    // ---- cooperative A build: wave handles pixels [wave*4, wave*4+4) ----
    #pragma unroll
    for (int pp = 0; pp < 4; ++pp) {
      int p = wave * 4 + pp;
      const float* o9 = off + (size_t)(P0 + p) * 16;
      #pragma unroll
      for (int k = 0; k < 3; ++k) {
        float dy  = o9[k], dx = o9[3 + k];
        float msk = sigm(o9[6 + k]);
        float py = dy + (float)y        + (isCol ? (float)(k - 1) : 0.0f);
        float px = dx + (float)(x0 + p) + (isCol ? 0.0f : (float)(k - 1));
        float fy = floorf(py), fx = floorf(px);
        int   yi = (int)fy,    xi = (int)fx;
        float ty = py - fy,    tx = px - fx;
        float w00 = (1.f - ty) * (1.f - tx), w01 = (1.f - ty) * tx;
        float w10 = ty * (1.f - tx),         w11 = ty * tx;
        #pragma unroll
        for (int cc = 0; cc < 2; ++cc) {
          int c = lane + cc * 32;                // 32 lanes x2 = 64 contiguous ch
          float v = w00 * fetch(yi,     xi,     c)
                  + w01 * fetch(yi,     xi + 1, c)
                  + w10 * fetch(yi + 1, xi,     c)
                  + w11 * fetch(yi + 1, xi + 1, c);
          A[p * 192 + k * 64 + c] = (_Float16)(v * msk);
        }
      }
    }
    __syncthreads();

    // ---- GEMM: [16 x 192] * [192 x 16] for this wave's N-tile ----
    v8f acc = {};
    #pragma unroll
    for (int ks = 0; ks < 6; ++ks) {
      v16h a;
      {
        v8h lo = *(const v8h*)&A[colx * 192 + ks * 32 + hi * 8];
        v8h up = *(const v8h*)&A[colx * 192 + ks * 32 + 16 + hi * 8];
        #pragma unroll
        for (int e = 0; e < 8; ++e) { a[e] = lo[e]; a[8 + e] = up[e]; }
      }
      v16h bm;
      #pragma unroll
      for (int j = 0; j < 16; ++j) {
        int K = ks * 32 + (j >> 3) * 16 + hi * 8 + (j & 7);
        int t = K / 64, c = K % 64;
        bm[j] = (_Float16)w[(size_t)(n * 64 + c) * 3 + t];   // OIHW: w[o][c][t]
      }
      acc = wmma16(a, bm, acc);
    }
    float bias = isCol ? bc[n] : br[n];
    #pragma unroll
    for (int r = 0; r < 8; ++r) acc[r] += bias;
    dacc[conv] = acc;
    __syncthreads();                           // A buffer reused by next conv
  }

  // ---- combine: out = 2*p_low + (sig(d_col)+sig(d_row)) * p_high (NCHW) ----
  #pragma unroll
  for (int r = 0; r < 8; ++r) {
    int m = r + hi * 8;
    size_t pix = (size_t)(P0 + m);
    float pl = plo[pix * 64 + n];
    float ph = phi[pix * 64 + n];
    float v  = 2.0f * pl + (sigm(dacc[0][r]) + sigm(dacc[1][r])) * ph;
    out[((size_t)b * 64 + n) * HW_ + (size_t)(s0 + m)] = v;
  }
}

// ---------------------------------------------------------------------------
extern "C" void kernel_launch(void* const* d_in, const int* in_sizes, int n_in,
                              void* d_out, int out_size, void* d_ws, size_t ws_size,
                              hipStream_t stream) {
  (void)in_sizes; (void)n_in; (void)out_size; (void)ws_size;

  const float* a_low  = (const float*)d_in[0];
  const float* a_high = (const float*)d_in[1];
  const float* pre_w1 = (const float*)d_in[2];
  const float* pre_b1 = (const float*)d_in[3];
  const float* pre_w2 = (const float*)d_in[4];
  const float* pre_b2 = (const float*)d_in[5];
  const float* offc_w = (const float*)d_in[6];
  const float* offc_b = (const float*)d_in[7];
  const float* wc     = (const float*)d_in[8];
  const float* bc     = (const float*)d_in[9];
  const float* offr_w = (const float*)d_in[10];
  const float* offr_b = (const float*)d_in[11];
  const float* wr     = (const float*)d_in[12];
  const float* br     = (const float*)d_in[13];

  // workspace: p_low NHWC (64MB), p_high NHWC (64MB), offsets 2x16MB = 160MB
  float* plo  = (float*)d_ws;
  float* phi  = plo  + (size_t)NPIX * 64;
  float* offc = phi  + (size_t)NPIX * 64;
  float* offr = offc + (size_t)NPIX * 16;

  k_pre   <<<dim3(NT / 4, 2), 128, 0, stream>>>(a_low, a_high, pre_w1, pre_b1,
                                                pre_w2, pre_b2, plo, phi);
  k_off   <<<dim3(NT / 4, 2), 128, 0, stream>>>(plo, offc_w, offc_b, offr_w, offr_b,
                                                offc, offr);
  k_deform<<<dim3(NT),        128, 0, stream>>>(plo, phi, offc, offr,
                                                wc, bc, wr, br, (float*)d_out);
}
